// Encoder_83528523972689
// MI455X (gfx1250) — matrix operand order repaired and checked by reading, not compiled, not run on hardware
//
#include <hip/hip_runtime.h>

// ---------------------------------------------------------------------------
// LSTM encoder for MI455X (gfx1250, wave32, WMMA + TDM).
//   B=32, T=512, E=512, H=1024, gates 4H=4096.
// Phase 0: pack W_ih/W_hh (f32 -> bf16, native cvt) into WMMA B-fragment
//          layout, zero h, reset the grid barrier counter.
// Phase 1: x_proj = bf16(emb[x]) @ bf16(W_ih)^T + (b_ih+b_hh).  The 32x512
//          A-tile is gathered+converted once per block into LDS, fragments
//          come from ds_load_b128, math is v_wmma_f32_16x16x32_bf16.
// Phase 2: persistent recurrence, 64 WGs x 256 threads.  Each WG owns 16
//          hidden units (all 4 gate blocks), keeps its 128 KB W_hh slice in
//          LDS, and pulls the shared 64 KB bf16 h into LDS each step with a
//          single TENSOR_LOAD_TO_LDS (TDM) op, synced via s_wait_tensorcnt.
// ---------------------------------------------------------------------------

#define BATCH 32
#define TLEN  512
#define EMBD  512
#define HIDD  1024
#define G4    4096

typedef __attribute__((ext_vector_type(16))) __bf16 v16bf;
typedef __attribute__((ext_vector_type(8)))  float  v8f;
typedef __attribute__((ext_vector_type(4)))  unsigned int v4u;
typedef __attribute__((ext_vector_type(8)))  unsigned int v8u;

union Frag  { v16bf v; uint4 q[2]; };
union Pack8 { uint4 q; __bf16 b[8]; };

__device__ __forceinline__ float sigm(float x) {
  return 1.0f / (1.0f + __expf(-x));
}

// ---- workspace layout (bytes) ---------------------------------------------
#define WIH_BYTES   ((size_t)G4 * EMBD * 2)                 //  4 MB packed bf16
#define WHH_BYTES   ((size_t)G4 * HIDD * 2)                 //  8 MB packed bf16
#define HBF_BYTES   ((size_t)BATCH * HIDD * 2)              // 64 KB bf16 h
#define OFF_WIH     ((size_t)0)
#define OFF_WHH     (OFF_WIH + WIH_BYTES)
#define OFF_HBF     (OFF_WHH + WHH_BYTES)
#define OFF_CTR     (OFF_HBF + HBF_BYTES)
#define OFF_XPROJ   (OFF_CTR + 256)
// x_proj: 16384 * 4096 * 4 = 256 MB

// ---------------------------------------------------------------------------
// Phase 0: pack one 32-byte lane-chunk of a B fragment per work item.
// B fragment as packed here (16-bit 32x16, wave32; A uses the same K order): lanes 0-15 hold N=lane, K=0..15;
// lanes 16-31 hold N=lane-16, K=16..31  => a lane's 16 bf16 are K-contiguous.
// Fragment order: frag = ntile*KT + kt ; lane chunk at (frag*32+lane)*32 B.
// ---------------------------------------------------------------------------
__device__ __forceinline__ void pack_unit(const float* __restrict__ W,
                                          unsigned short* __restrict__ dst,
                                          int K, int KT, int unit) {
  int frag = unit >> 5, lane = unit & 31;
  int ntile = frag / KT, kt = frag - ntile * KT;
  int row = ntile * 16 + (lane & 15);
  int kk  = kt * 32 + ((lane >> 4) << 4);
  const float4* s = (const float4*)(W + (size_t)row * K + kk);
  uint4* d = (uint4*)(dst + (size_t)unit * 16);
#pragma unroll
  for (int h = 0; h < 2; ++h) {
    float4 f0 = s[2 * h], f1 = s[2 * h + 1];
    Pack8 u;
    u.b[0] = (__bf16)f0.x; u.b[1] = (__bf16)f0.y;
    u.b[2] = (__bf16)f0.z; u.b[3] = (__bf16)f0.w;
    u.b[4] = (__bf16)f1.x; u.b[5] = (__bf16)f1.y;
    u.b[6] = (__bf16)f1.z; u.b[7] = (__bf16)f1.w;
    d[h] = u.q;
  }
}

__global__ void prep_kernel(const float* __restrict__ W_ih,
                            const float* __restrict__ W_hh,
                            unsigned short* __restrict__ wih_p,
                            unsigned short* __restrict__ whh_p,
                            unsigned int*  __restrict__ hbf_zero,
                            unsigned long long* __restrict__ ctr) {
  int n = blockIdx.x * blockDim.x + threadIdx.x;
  const int NIH = (G4 / 16) * (EMBD / 32) * 32;   // 131072 units
  const int NHH = (G4 / 16) * (HIDD / 32) * 32;   // 262144 units
  const int NHZ = (BATCH * HIDD) / 2;             // 16384 dwords of bf16 h
  if (n < NIH) { pack_unit(W_ih, wih_p, EMBD, EMBD / 32, n); return; }
  n -= NIH;
  if (n < NHH) { pack_unit(W_hh, whh_p, HIDD, HIDD / 32, n); return; }
  n -= NHH;
  if (n < NHZ) { hbf_zero[n] = 0u; return; }
  n -= NHZ;
  if (n == 0)  { *ctr = 0ull; }
}

// ---------------------------------------------------------------------------
// Phase 1: x_proj GEMM.  Block = 256 thr (8 waves) -> 32(M) x 64(N) tile.
// The block stages its 32x512 bf16 A-tile (gathered via token ids) in LDS
// once; each wave then reads A fragments with ds_load_b128.  K=512 -> 16
// WMMAs per wave.
// ---------------------------------------------------------------------------
__global__ void __launch_bounds__(256)
xproj_kernel(const int* __restrict__ x,
             const float* __restrict__ emb,
             const unsigned short* __restrict__ wih_p,
             const float* __restrict__ b_ih,
             const float* __restrict__ b_hh,
             float* __restrict__ xproj) {
  __shared__ unsigned short a_lds[32 * EMBD];   // 32 KB bf16 A-tile

  const int tid  = threadIdx.x;
  const int lane = tid & 31, wv = tid >> 5;
  const int mhalf = wv & 1, nt = wv >> 1;
  const int ntile = blockIdx.y * 4 + nt;
  const int mbase = blockIdx.x * 32 + mhalf * 16;

  // ---- cooperative gather + f32->bf16 convert of the A-tile ----
  for (int c = tid; c < (32 * EMBD) / 8; c += 256) {   // 8 elems per chunk
    int row = c >> 6, k8 = (c & 63) * 8;
    int tok = x[blockIdx.x * 32 + row];
    const float4* s = (const float4*)(emb + (size_t)tok * EMBD + k8);
    float4 f0 = s[0], f1 = s[1];
    Pack8 u;
    u.b[0] = (__bf16)f0.x; u.b[1] = (__bf16)f0.y;
    u.b[2] = (__bf16)f0.z; u.b[3] = (__bf16)f0.w;
    u.b[4] = (__bf16)f1.x; u.b[5] = (__bf16)f1.y;
    u.b[6] = (__bf16)f1.z; u.b[7] = (__bf16)f1.w;
    *(uint4*)(a_lds + row * EMBD + k8) = u.q;
  }
  __syncthreads();

  const int g = ntile * 16 + (lane & 15);
  float bias = b_ih[g] + b_hh[g];
  v8f acc;
#pragma unroll
  for (int r = 0; r < 8; ++r) acc[r] = bias;   // bias depends only on N

  const int run0 = (lane >> 4) << 4;           // A-fragment K-run start
  const unsigned short* arow = a_lds + (mhalf * 16 + (lane & 15)) * EMBD;

#pragma unroll
  for (int kt = 0; kt < EMBD / 32; ++kt) {
    Frag a, b;
    a.q[0] = *(const uint4*)(arow + kt * 32 + run0);
    a.q[1] = *(const uint4*)(arow + kt * 32 + run0 + 8);
    const uint4* bp = (const uint4*)(wih_p + (size_t)(ntile * (EMBD / 32) + kt) * 512) + lane * 2;
    b.q[0] = bp[0]; b.q[1] = bp[1];
    acc = __builtin_amdgcn_wmma_f32_16x16x32_bf16(false, a.v, false, b.v,
                                                  (short)0, acc, false, false);
  }
#pragma unroll
  for (int r = 0; r < 8; ++r) {
    int row = mbase + r + ((lane >> 4) << 3);  // C/D layout: M = r + 8*(lane>=16)
    xproj[(size_t)row * G4 + g] = acc[r];
  }
}

// ---------------------------------------------------------------------------
// Phase 2: persistent recurrence.  64 WGs x 256 threads, dynamic LDS:
//   [0,128K)    W_hh slice (4 gate ntiles x 32 kt x 32 lanes x 32 B)
//   [128K,192K) staged bf16 h (32 x 1024), filled by TENSOR_LOAD_TO_LDS
//   [+8K)       gate exchange (4 x 32 x 16 f32)
//   [+2K)       cell state c (32 x 16 f32)
// ---------------------------------------------------------------------------
#define LDS_WHH   (131072)
#define LDS_H     (65536)
#define LDS_GATES (8192)
#define LDS_C     (2048)
#define LDS_TOTAL (LDS_WHH + LDS_H + LDS_GATES + LDS_C)

__global__ void __launch_bounds__(256, 1)
lstm_kernel(const float* __restrict__ xproj,
            const unsigned short* __restrict__ whh_p,
            unsigned short* __restrict__ hbf,
            float* __restrict__ out,
            unsigned long long* __restrict__ ctr) {
  extern __shared__ char smem[];
  unsigned short* whh_lds = (unsigned short*)smem;
  unsigned short* h_lds   = (unsigned short*)(smem + LDS_WHH);
  float* gates            = (float*)(smem + LDS_WHH + LDS_H);
  float* c_lds            = (float*)(smem + LDS_WHH + LDS_H + LDS_GATES);

  const int tid  = threadIdx.x;
  const int wg   = blockIdx.x;          // owns hidden units [wg*16, wg*16+16)
  const int lane = tid & 31, wv = tid >> 5;
  const int mhalf = wv & 1;             // batch half: rows 0-15 / 16-31
  const int gt    = wv >> 1;            // gate block: 0=i 1=f 2=g 3=o
  const int NWG   = gridDim.x;

  // ---- Tensor DMA descriptor (D#) for the per-step h broadcast:
  //      flat 1-D tile, 8192 x 8-byte units = 64 KB, global hbf -> LDS h_lds.
  //      Built once from kernarg-uniform values (lives in SGPRs).
  unsigned long long gaddr = (unsigned long long)(uintptr_t)hbf;
  unsigned int ldsoff = (unsigned int)(uintptr_t)h_lds;   // low 32 = LDS byte addr
  v4u g0;
  g0[0] = 1u;                                             // count=1 (valid user D#)
  g0[1] = ldsoff;                                         // lds_addr
  g0[2] = (unsigned int)(gaddr & 0xffffffffu);            // global_addr[31:0]
  g0[3] = (unsigned int)((gaddr >> 32) & 0x01ffffffu)     // global_addr[56:32]
        | 0x80000000u;                                    // type=2 ("image")
  v8u g1;
  g1[0] = 3u << 16;                                       // data_size=3 (8B), mask=0
  g1[1] = (8192u & 0xffffu) << 16;                        // tensor_dim0[15:0]
  g1[2] = (8192u >> 16) | (1u << 16);                     // tensor_dim0[31:16], tensor_dim1=1
  g1[3] = (8192u & 0xffffu) << 16;                        // tile_dim0=8192
  g1[4] = 0u;                                             // tile_dim1/2 unused
  g1[5] = 8192u;                                          // tensor_dim0_stride lo
  g1[6] = 0u;
  g1[7] = 0u;

  // ---- load this WG's W_hh fragments into LDS (4 x 32 KB) ----
  {
    const uint4* wsrc = (const uint4*)whh_p;
    uint4* wdst = (uint4*)whh_lds;
#pragma unroll
    for (int g2 = 0; g2 < 4; ++g2) {
      int ntile = g2 * 64 + wg;                       // global gate ntile
      const uint4* s = wsrc + (size_t)ntile * 2048;   // 32 KB slice
      uint4* d = wdst + g2 * 2048;
      for (int i = tid; i < 2048; i += 256) d[i] = s[i];
    }
  }
  for (int i = tid; i < 512; i += 256) c_lds[i] = 0.0f;
  __syncthreads();

  const int run0 = (lane >> 4) << 4;
  const int g    = gt * HIDD + wg * 16 + (lane & 15); // gate row (N) index

  for (int t = 0; t < TLEN; ++t) {
    // --- TDM: pull shared h (64 KB bf16) global -> LDS, one op per WG ---
    if (wv == 0) {
      asm volatile("tensor_load_to_lds %0, %1" :: "s"(g0), "s"(g1) : "memory");
      __builtin_amdgcn_s_wait_tensorcnt(0);
    }
    __syncthreads();

    // --- accumulator := x_proj gate tile (bias already folded in) ---
    v8f acc;
#pragma unroll
    for (int r = 0; r < 8; ++r) {
      int bm = mhalf * 16 + r + ((lane >> 4) << 3);
      acc[r] = xproj[((size_t)bm * TLEN + t) * G4 + g];
      if (t + 1 < TLEN)   // pull next timestep's slice toward L2 early
        __builtin_prefetch(&xproj[((size_t)bm * TLEN + (t + 1)) * G4 + g], 0, 1);
    }

    // --- 32 chained WMMAs: gates += h @ W_hh^T (bf16 in, f32 acc) ---
    const unsigned short* hrow = h_lds + (mhalf * 16 + (lane & 15)) * HIDD;
    const uint4* bbase = (const uint4*)whh_lds + (size_t)gt * 2048 + lane * 2;
#pragma unroll
    for (int kt = 0; kt < 32; ++kt) {
      Frag a, b;
      a.q[0] = *(const uint4*)(hrow + kt * 32 + run0);
      a.q[1] = *(const uint4*)(hrow + kt * 32 + run0 + 8);
      const uint4* bp = bbase + kt * 64;
      b.q[0] = bp[0]; b.q[1] = bp[1];
      acc = __builtin_amdgcn_wmma_f32_16x16x32_bf16(false, a.v, false, b.v,
                                                    (short)0, acc, false, false);
    }

    // --- per-gate nonlinearity, exchange through LDS ---
#pragma unroll
    for (int r = 0; r < 8; ++r) {
      float v = acc[r];
      v = (gt == 2) ? tanhf(v) : sigm(v);
      int bm = mhalf * 16 + r + ((lane >> 4) << 3);
      gates[(gt * 32 + bm) * 16 + (lane & 15)] = v;
    }
    __syncthreads();

    // --- cell update for this WG's 32x16 (batch x hidden) slice ---
    for (int idx = tid; idx < 512; idx += 256) {
      int m = idx >> 4, n = idx & 15;
      float iv = gates[(0 * 32 + m) * 16 + n];
      float fv = gates[(1 * 32 + m) * 16 + n];
      float gv = gates[(2 * 32 + m) * 16 + n];
      float ov = gates[(3 * 32 + m) * 16 + n];
      float c  = fv * c_lds[idx] + iv * gv;
      c_lds[idx] = c;
      float h = ov * tanhf(c);
      int hcol = wg * 16 + n;
      unsigned short hb;
      { __bf16 t16 = (__bf16)h; hb = *(unsigned short*)&t16; }
      hbf[m * HIDD + hcol] = hb;                               // republish bf16 h
      out[((size_t)m * TLEN + t) * HIDD + hcol] = h;           // encoder_outputs
      if (t == TLEN - 1) {
        out[(size_t)BATCH * TLEN * HIDD + (size_t)m * HIDD + hcol] = h;  // state_h
        out[(size_t)BATCH * TLEN * HIDD + (size_t)BATCH * HIDD
            + (size_t)m * HIDD + hcol] = c;                               // state_c
      }
    }

    // --- device-wide step barrier (persistent grid, monotone counter) ---
    __threadfence();
    __syncthreads();
    if (tid == 0) {
      __hip_atomic_fetch_add(ctr, 1ull, __ATOMIC_RELEASE, __HIP_MEMORY_SCOPE_AGENT);
      unsigned long long tgt = (unsigned long long)(t + 1) * (unsigned long long)NWG;
      while (__hip_atomic_load(ctr, __ATOMIC_ACQUIRE, __HIP_MEMORY_SCOPE_AGENT) < tgt)
        __builtin_amdgcn_s_sleep(1);
    }
    __syncthreads();
  }
}

// ---------------------------------------------------------------------------
extern "C" void kernel_launch(void* const* d_in, const int* in_sizes, int n_in,
                              void* d_out, int out_size, void* d_ws, size_t ws_size,
                              hipStream_t stream) {
  const int*   x    = (const int*)d_in[0];
  const float* emb  = (const float*)d_in[1];
  const float* W_ih = (const float*)d_in[2];
  const float* W_hh = (const float*)d_in[3];
  const float* b_ih = (const float*)d_in[4];
  const float* b_hh = (const float*)d_in[5];
  float* out = (float*)d_out;

  char* ws = (char*)d_ws;
  unsigned short* wih_p = (unsigned short*)(ws + OFF_WIH);
  unsigned short* whh_p = (unsigned short*)(ws + OFF_WHH);
  unsigned short* hbf   = (unsigned short*)(ws + OFF_HBF);
  unsigned long long* ctr = (unsigned long long*)(ws + OFF_CTR);
  float* xproj          = (float*)(ws + OFF_XPROJ);

  // Phase 0: weight packing + state/counter reset (deterministic every call).
  {
    int total = 131072 + 262144 + 16384 + 1;
    prep_kernel<<<(total + 255) / 256, 256, 0, stream>>>(
        W_ih, W_hh, wih_p, whh_p, (unsigned int*)hbf, ctr);
  }
  // Phase 1: input projection GEMM (16384 x 4096, K=512).
  {
    dim3 grid((BATCH * TLEN) / 32, G4 / 64);   // (512, 64)
    xproj_kernel<<<grid, 256, 0, stream>>>(x, emb, wih_p, b_ih, b_hh, xproj);
  }
  // Phase 2: persistent LSTM recurrence (64 WGs, one per 16 hidden units).
  lstm_kernel<<<64, 256, LDS_TOTAL, stream>>>(xproj, whh_p, hbf, out, ctr);
}
